// PSN_76716705841952
// MI455X (gfx1250) — matrix operand order repaired and checked by reading, not compiled, not run on hardware
//
#include <hip/hip_runtime.h>
#include <hip/hip_bf16.h>
#include <math.h>

typedef __attribute__((ext_vector_type(16))) _Float16     v16h;
typedef __attribute__((ext_vector_type(8)))  float        v8f;
typedef __attribute__((ext_vector_type(4)))  unsigned int v4u;
typedef __attribute__((ext_vector_type(8)))  int          v8i;
typedef __attribute__((ext_vector_type(4)))  int          v4i;

#define BB   4
#define MM   8192
#define DD   64
#define SS   1024
#define H0L  32
#define H1L  128
#define EPSV 1e-5f
#define TAUV 0.1f

// output offsets (in floats), concatenated in reference return order
#define OFF_SP 0          // sampled_points  [4,1024,3]
#define OFF_GP 12288      // grouped_points  [4,1024,32,3]
#define OFF_SF 405504     // sampled_feature [4,1024,64]
#define OFF_GF 667648     // grouped_feature [4,1024,32,64]
#define OFF_QG 9056256    // Qg              [4,1024,8192]

// ---- CDNA5 data-mover selection (guarded so the file always compiles) ------
#if defined(__has_builtin)
# if __has_builtin(__builtin_amdgcn_tensor_load_to_lds)
#  define HAVE_TDM 1
# elif __has_builtin(__builtin_amdgcn_load_to_lds)
#  define HAVE_ASYNC_LDS 1
# endif
#endif

__device__ __forceinline__ unsigned lds_offset_of(const void* p) {
    // AS3 pointer value == byte offset within the wave's LDS allocation
    return (unsigned)(unsigned long long)
        (__attribute__((address_space(3))) const void*)p;
}

#ifdef HAVE_TDM
// Issue one TDM descriptor: copy `nbytes/4` dwords from gsrc -> LDS offset.
// D# packing per CDNA5 ISA (1-D tile, no
// cluster multicast, no padding, no gather, no iterate).
__device__ __forceinline__ void tdm_load_tile(const void* gsrc, unsigned lds_off,
                                              unsigned ndwords) {
    const unsigned long long ga = (unsigned long long)gsrc;
    v4u g0;
    g0[0] = 1u;                                   // count=1 (valid), user mode
    g0[1] = lds_off;                              // lds_addr (bytes)
    g0[2] = (unsigned)(ga & 0xffffffffu);         // global_addr[31:0]
    g0[3] = (unsigned)((ga >> 32) & 0x01ffffffu)  // global_addr[56:32]
          | 0x80000000u;                          // type=2 ("image")
    v8i g1;
    g1[0] = (int)(2u << 16);                      // wg_mask=0, data_size=4B
    g1[1] = (int)(ndwords << 16);                 // tensor_dim0[15:0] @127? no: bits63:48
    g1[2] = (int)(1u << 16);                      // tensor_dim0 hi=0 | tensor_dim1=1
    g1[3] = (int)(ndwords << 16);                 // tensor_dim1 hi=0 | tile_dim0
    g1[4] = 0;                                    // tile_dim1=0, tile_dim2=0 (unused)
    g1[5] = (int)ndwords;                         // tensor_dim0_stride[31:0]
    g1[6] = 0;                                    // stride hi | tensor_dim1_stride lo
    g1[7] = 0;
    const v4i z4 = { 0, 0, 0, 0 };
# if __clang_major__ >= 23
    const v8i z8 = { 0, 0, 0, 0, 0, 0, 0, 0 };
    __builtin_amdgcn_tensor_load_to_lds(g0, g1, z4, z4, z8, 0);
# else
    __builtin_amdgcn_tensor_load_to_lds(g0, g1, z4, z4, 0);
# endif
}
#endif

// Stage one 4KB B tile into LDS. TDM path: wave 0 issues a single DMA.
// Fallbacks: per-thread async load-to-LDS, else plain 16B copies.
__device__ __forceinline__ void stage_tile(const _Float16* __restrict__ src,
                                           _Float16* __restrict__ dst,
                                           int t, int wave) {
#if defined(HAVE_TDM)
    if (wave == 0) tdm_load_tile(src, lds_offset_of(dst), 1024u);
#elif defined(HAVE_ASYNC_LDS)
    __builtin_amdgcn_load_to_lds((void*)(src + t * 8),
        (__attribute__((address_space(3))) void*)(dst + t * 8), 16, 0, 0);
#else
    const float4 tmp = *(const float4*)(src + t * 8);
    *(float4*)(dst + t * 8) = tmp;
#endif
}

__device__ __forceinline__ void stage_wait(int wave) {
#if defined(HAVE_TDM)
    if (wave == 0) __builtin_amdgcn_s_wait_tensorcnt(0);
#elif defined(HAVE_ASYNC_LDS)
    asm volatile("s_wait_asynccnt 0x0" ::: "memory");
#endif
    (void)wave;
}

// ---------------------------------------------------------------------------
// Kernel 1: repack w2 (f32 [128][1024]) -> f16 in WMMA B-operand order:
// packed[((ct*4+kk)*32 + lane)*16 + j] = w2[kk*32 + (lane>>4)*16 + j][ct*16 + (lane&15)]
// ---------------------------------------------------------------------------
__global__ __launch_bounds__(256)
void pack_w2_f16(const float* __restrict__ w2, _Float16* __restrict__ w2h) {
    int tid  = blockIdx.x * 256 + threadIdx.x;   // 131072 total
    int j    = tid & 15;
    int lane = (tid >> 4) & 31;
    int kk   = (tid >> 9) & 3;
    int ct   = tid >> 11;
    int k    = kk * 32 + (lane >> 4) * 16 + j;
    int s    = ct * 16 + (lane & 15);
    w2h[tid] = (_Float16)w2[k * SS + s];
}

// ---------------------------------------------------------------------------
// Kernel 2: per-point MLP (5->32->128, BN+ReLU) then WMMA GEMM h1[128x128] @
// w2[128x1024], sigmoid, store Q transposed into the Qg output region.
// 256 blocks (b * 64 m-tiles of 128 points), 256 threads (8 waves, one 16-row
// tile each). B tiles (4KB per 16-col tile) are double-buffered in LDS via the
// CDNA5 Tensor Data Mover and shared by all 8 waves.
// ---------------------------------------------------------------------------
__global__ __launch_bounds__(256)
void mlp_gemm_kernel(const float* __restrict__ coord,
                     const float* __restrict__ w0,
                     const float* __restrict__ g0, const float* __restrict__ b0,
                     const float* __restrict__ mu0, const float* __restrict__ va0,
                     const float* __restrict__ w1,
                     const float* __restrict__ g1, const float* __restrict__ b1,
                     const float* __restrict__ mu1, const float* __restrict__ va1,
                     const _Float16* __restrict__ w2h,
                     float* __restrict__ Qout) {
    __shared__ float    sh0[128][H0L];    // 16 KB  h0 (f32)
    __shared__ _Float16 sh1[128][H1L];    // 32 KB  h1 (f16, A-operand source)
    __shared__ _Float16 shB[2][2048];     //  8 KB  double-buffered B tile
    const int t = threadIdx.x;
    const int b = blockIdx.x >> 6;
    const int m_base = (blockIdx.x & 63) << 7;

    // Stage A: spherical features + layer 0 (threads 0..127, one point each)
    if (t < 128) {
        const int m = m_base + t;
        const float x  = coord[(b * MM + m) * 3 + 0];
        const float y  = coord[(b * MM + m) * 3 + 1];
        const float zc = coord[(b * MM + m) * 3 + 2];
        const float r  = sqrtf(x * x + y * y + zc * zc);
        const float th = acosf(zc / r);
        const float fi = atan2f(y, x);
        const float c2[5] = { x, y, zc, th, fi };
        for (int j = 0; j < H0L; ++j) {
            float acc = 0.f;
            for (int i = 0; i < 5; ++i) acc += c2[i] * w0[i * H0L + j];
            float v = (acc - mu0[j]) * rsqrtf(va0[j] + EPSV) * g0[j] + b0[j];
            sh0[t][j] = fmaxf(v, 0.f);
        }
    }
    __syncthreads();

    // Stage B: layer 1 (each thread: 1 point, 64 of 128 channels), store f16
    {
        const int p = t >> 1, j0 = (t & 1) << 6;
        for (int j = 0; j < 64; ++j) {
            const int jj = j0 + j;
            float acc = 0.f;
            for (int i = 0; i < H0L; ++i) acc += sh0[p][i] * w1[i * H1L + jj];
            float v = (acc - mu1[jj]) * rsqrtf(va1[jj] + EPSV) * g1[jj] + b1[jj];
            sh1[p][jj] = (_Float16)fmaxf(v, 0.f);
        }
    }
    __syncthreads();

    // Stage C: WMMA GEMM. wave w owns row tile rt = w (rows w*16..w*16+15).
    const int wave = t >> 5, lane = t & 31;
    const int rt = wave;
    const int hi = lane >> 4, c = lane & 15;

    // Pack A (16x32 f16 per k-step): lane<16 row=lane K={0..15};
    // lane>=16 same row K={16..31}
    v16h a[4];
    const int row = rt * 16 + c;
    for (int kk = 0; kk < 4; ++kk)
        for (int j = 0; j < 16; ++j) {
            const int k0 = hi * 16 + j;
            a[kk][j] = sh1[row][kk * 32 + k0];
        }

    // prologue: stage B tile for ct=0 into buffer 0
    stage_tile(w2h, &shB[0][0], t, wave);
    stage_wait(wave);
    __syncthreads();

    int cur = 0;
    for (int u = 0; u < 64; ++u) {
        const int nxt = cur ^ 1;
        if (u + 1 < 64)  // issue next tile's DMA; overlaps with WMMAs below
            stage_tile(w2h + (size_t)(u + 1) * 2048, &shB[nxt][0], t, wave);

        v8f acc = { 0.f, 0.f, 0.f, 0.f, 0.f, 0.f, 0.f, 0.f };
        for (int kk = 0; kk < 4; ++kk) {
            v16h bv = *(const v16h*)(&shB[cur][kk * 512 + lane * 16]);
            acc = __builtin_amdgcn_wmma_f32_16x16x32_f16(
                false, a[kk], false, bv, (short)0, acc, false, false);
        }
        // D layout: VGPR r -> m = rt*16 + hi*8 + r, col s = ct*16 + (lane&15)
        const int s = u * 16 + c;
        float* q = Qout + (size_t)(b * SS + s) * MM + m_base + rt * 16 + hi * 8;
        for (int r = 0; r < 8; ++r)
            q[r] = __builtin_amdgcn_rcpf(1.0f + __expf(-acc[r]));  // sigmoid

        if (u + 1 < 64) stage_wait(wave);
        __syncthreads();
        cur = nxt;
    }
}

// ---------------------------------------------------------------------------
// Kernel 3: top-32 per (b,s) row of Q (value desc, index asc on ties), then
// gather grouped_points / grouped_feature. One block per row.
// ---------------------------------------------------------------------------
__global__ __launch_bounds__(256)
void topk_kernel(const float* __restrict__ Q,
                 const float* __restrict__ coord,
                 const float* __restrict__ feat,
                 float* __restrict__ gp, float* __restrict__ gf) {
    __shared__ float rv[256];
    __shared__ int   ri[256];
    __shared__ int   sel[32];
    const int t = threadIdx.x;
    const int b = blockIdx.x >> 10;
    const int s = blockIdx.x & 1023;
    const float* qrow = Q + (size_t)(b * SS + s) * MM;

    float vals[32];
    for (int j = 0; j < 32; ++j) vals[j] = qrow[t + (j << 8)];

    unsigned taken = 0u;
    for (int k = 0; k < 32; ++k) {
        float bv = -1.0f; int bm = 0x7fffffff;   // Q in (0,1): -1 is a safe sentinel
        for (int j = 0; j < 32; ++j) {
            if (taken & (1u << j)) continue;
            const float v = vals[j]; const int m = t + (j << 8);
            if (v > bv || (v == bv && m < bm)) { bv = v; bm = m; }
        }
        rv[t] = bv; ri[t] = bm;
        __syncthreads();
        for (int off = 128; off > 0; off >>= 1) {
            if (t < off) {
                const float ov = rv[t + off]; const int om = ri[t + off];
                if (ov > rv[t] || (ov == rv[t] && om < ri[t])) { rv[t] = ov; ri[t] = om; }
            }
            __syncthreads();
        }
        const int win = ri[0];
        if (t == 0) sel[k] = win;
        if ((win & 255) == t) taken |= (1u << (win >> 8));
        __syncthreads();
    }

    const size_t rowo = (size_t)(b * SS + s);
    for (int u = t; u < 96; u += 256) {
        const int n = u / 3, cc = u % 3;
        gp[(rowo * 32 + n) * 3 + cc] = coord[((size_t)b * MM + sel[n]) * 3 + cc];
    }
    for (int u = t; u < 2048; u += 256) {
        const int n = u >> 6, cc = u & 63;
        gf[(rowo * 32 + n) * 64 + cc] = feat[((size_t)b * MM + sel[n]) * 64 + cc];
    }
}

// ---------------------------------------------------------------------------
// Kernel 4: hard gumbel-softmax. Qg is exactly one scalar per row at argmax:
// qh = fl(fl(1 - y_hard) + y_hard), y_hard = 1/sum(exp(z - zmax)).
// Overwrites Q row with Qg; writes sampled_* and grouped_feature[:, :, 0, :].
// ---------------------------------------------------------------------------
__global__ __launch_bounds__(256)
void gumbel_kernel(float* __restrict__ Qg,         // in: Q, out: Qg (same region)
                   const float* __restrict__ gum,
                   const float* __restrict__ coord,
                   const float* __restrict__ feat,
                   float* __restrict__ sp, float* __restrict__ sf,
                   float* __restrict__ gf) {
    __shared__ float rv[256];
    __shared__ int   ri[256];
    const int t = threadIdx.x;
    const int b = blockIdx.x >> 10;
    const int s = blockIdx.x & 1023;
    float* qrow = Qg + (size_t)(b * SS + s) * MM;
    const float* grow = gum + (size_t)(b * SS + s) * MM;

    float z[32];
    for (int j = 0; j < 32; ++j) {
        const int m = t + (j << 8);
        z[j] = (qrow[m] + grow[m]) * (1.0f / TAUV);
    }
    // argmax (first index on ties, like jnp.argmax)
    float bv = -INFINITY; int bm = 0x7fffffff;
    for (int j = 0; j < 32; ++j) {
        const int m = t + (j << 8);
        if (z[j] > bv || (z[j] == bv && m < bm)) { bv = z[j]; bm = m; }
    }
    rv[t] = bv; ri[t] = bm;
    __syncthreads();
    for (int off = 128; off > 0; off >>= 1) {
        if (t < off) {
            const float ov = rv[t + off]; const int om = ri[t + off];
            if (ov > rv[t] || (ov == rv[t] && om < ri[t])) { rv[t] = ov; ri[t] = om; }
        }
        __syncthreads();
    }
    const float zmax = rv[0];
    const int   hard = ri[0];
    __syncthreads();

    float ls = 0.f;
    for (int j = 0; j < 32; ++j) ls += __expf(z[j] - zmax);
    rv[t] = ls;
    __syncthreads();
    for (int off = 128; off > 0; off >>= 1) {
        if (t < off) rv[t] += rv[t + off];
        __syncthreads();
    }
    const float yh = 1.0f / rv[0];        // exp(z_hard - zmax) == exp(0) == 1
    const float qh = (1.0f - yh) + yh;    // straight-through value at argmax

    for (int j = 0; j < 32; ++j) {
        const int m = t + (j << 8);
        qrow[m] = (m == hard) ? qh : 0.0f;  // (0 - y) + y == +0 exactly elsewhere
    }
    const size_t rowo = (size_t)(b * SS + s);
    if (t < 3)
        sp[rowo * 3 + t] = qh * coord[((size_t)b * MM + hard) * 3 + t];
    if (t < 64) {
        const float v = qh * feat[((size_t)b * MM + hard) * 64 + t];
        sf[rowo * 64 + t]        = v;
        gf[(rowo * 32) * 64 + t] = v;      // grouped_feature[:, :, 0, :]
    }
}

// ---------------------------------------------------------------------------
extern "C" void kernel_launch(void* const* d_in, const int* in_sizes, int n_in,
                              void* d_out, int out_size, void* d_ws, size_t ws_size,
                              hipStream_t stream) {
    (void)in_sizes; (void)n_in; (void)out_size; (void)ws_size;
    const float* coord = (const float*)d_in[0];
    const float* feat  = (const float*)d_in[1];
    const float* gum   = (const float*)d_in[2];
    const float* w0    = (const float*)d_in[3];
    const float* g0    = (const float*)d_in[4];
    const float* b0    = (const float*)d_in[5];
    const float* mu0   = (const float*)d_in[6];
    const float* va0   = (const float*)d_in[7];
    const float* w1    = (const float*)d_in[8];
    const float* g1    = (const float*)d_in[9];
    const float* b1    = (const float*)d_in[10];
    const float* mu1   = (const float*)d_in[11];
    const float* va1   = (const float*)d_in[12];
    const float* w2    = (const float*)d_in[13];

    float* out = (float*)d_out;
    float* sp = out + OFF_SP;
    float* gp = out + OFF_GP;
    float* sf = out + OFF_SF;
    float* gf = out + OFF_GF;
    float* qg = out + OFF_QG;
    _Float16* w2h = (_Float16*)d_ws;   // 256 KB packed f16 w2

    pack_w2_f16    <<<512,  256, 0, stream>>>(w2, w2h);
    mlp_gemm_kernel<<<256,  256, 0, stream>>>(coord, w0, g0, b0, mu0, va0,
                                              w1, g1, b1, mu1, va1, w2h, qg);
    topk_kernel    <<<4096, 256, 0, stream>>>(qg, coord, feat, gp, gf);
    gumbel_kernel  <<<4096, 256, 0, stream>>>(qg, gum, coord, feat, sp, sf, gf);
}
